// SelfAttentionLayer_79680233276145
// MI455X (gfx1250) — matrix operand order repaired and checked by reading, not compiled, not run on hardware
//
#include <hip/hip_runtime.h>
#include <hip/hip_bf16.h>

// ---------------------------------------------------------------------------
// CDNA5 (gfx1250) fused self-attention:
//   1) qkv_proj_kernel: Q/K/V = x @ W{Q,K,V}  (fp32 in, bf16 out, WMMA bf16)
//   2) flash_attn_kernel: online-softmax flash attention (WMMA bf16, fp32 acc)
//      with async (ASYNCcnt) K-tile staging into LDS.
// ---------------------------------------------------------------------------

typedef __attribute__((ext_vector_type(16))) __bf16 v16bf;
typedef __attribute__((ext_vector_type(8)))  __bf16 v8bf;
typedef __attribute__((ext_vector_type(4)))  __bf16 v4bf;
typedef __attribute__((ext_vector_type(8)))  float  v8f;
typedef __attribute__((ext_vector_type(4)))  float  v4f;

#define WMMA_BF16(A, B, C) \
  __builtin_amdgcn_wmma_f32_16x16x32_bf16(false, (A), false, (B), (short)0, (C), false, false)

static __device__ inline v16bf frag_pair(const __bf16* p0, const __bf16* p1) {
  v8bf lo = *(const v8bf*)p0;
  v8bf hi = *(const v8bf*)p1;
  v16bf r;
#pragma unroll
  for (int i = 0; i < 8; ++i) { r[i] = lo[i]; r[i + 8] = hi[i]; }
  return r;
}

// A-matrix frag (16x32 bf16): lane = 16h+m; elems 0..7 <-> k=8h..8h+7,
// elems 8..15 <-> k=16+8h..16+8h+7.  `row` points at (row m, k=0).
static __device__ inline v16bf load_a(const __bf16* row, int h) {
  return frag_pair(row + 8 * h, row + 16 + 8 * h);
}
// B-matrix frag (32x16 bf16): lane = 16h+n; elems 0..7 <-> k=8h..8h+7,
// elems 8..15 <-> k=16+8h..16+8h+7.  `coln` points at (column n, k=0).
static __device__ inline v16bf load_b(const __bf16* coln, int h) {
  return frag_pair(coln + 8 * h, coln + 16 + 8 * h);
}

// Async 16B copy global -> LDS (tracked by ASYNCcnt, no VGPR round trip).
// GVS mode: mem addr = SGPR64 base + per-lane 32-bit byte offset.
static __device__ inline void async_copy_b128(unsigned lds_addr, unsigned gvoff,
                                              unsigned long long gbase) {
  asm volatile("global_load_async_to_lds_b128 %0, %1, %2"
               :: "v"(lds_addr), "v"(gvoff), "s"(gbase) : "memory");
}
static __device__ inline void wait_asynccnt0() {
  asm volatile("s_wait_asynccnt 0x0" ::: "memory");
}

// ---------------------------------------------------------------------------
// QKV projection: [16384,512] x [512,512] -> bf16 [16384,512]; scale folded in.
// Block: 256 thr (8 waves), tile 128x128, K-step 32. Wave = 32(M) x 64(N).
// ---------------------------------------------------------------------------
__global__ __launch_bounds__(256)
void qkv_proj_kernel(const float* __restrict__ x, const float* __restrict__ W,
                     __bf16* __restrict__ out, float scale) {
  constexpr int K = 512, N = 512;
  __shared__ __bf16 As[128][40];  // x tile (row-major), padded
  __shared__ __bf16 Wt[128][40];  // W tile transposed: Wt[n][k], padded

  const int tid  = threadIdx.x;
  const int lane = tid & 31, wv = tid >> 5;
  const int h = lane >> 4, ln = lane & 15;
  const int mW = wv & 3;        // 4 M-groups of 32 rows
  const int nW = wv >> 2;       // 2 N-groups of 64 cols
  const int bM = blockIdx.x * 128;
  const int bN = blockIdx.y * 128;

  v8f acc[2][4];
#pragma unroll
  for (int mt = 0; mt < 2; ++mt)
#pragma unroll
    for (int nt = 0; nt < 4; ++nt)
#pragma unroll
      for (int r = 0; r < 8; ++r) acc[mt][nt][r] = 0.0f;

  for (int kk = 0; kk < K; kk += 32) {
    {  // stage x tile (128x32 f32 -> bf16)
      int r = tid >> 1, c0 = (tid & 1) * 16;
      const float* src = x + (size_t)(bM + r) * K + kk + c0;
#pragma unroll
      for (int j = 0; j < 4; ++j) {
        v4f v = *(const v4f*)(src + j * 4);
#pragma unroll
        for (int i = 0; i < 4; ++i) As[r][c0 + j * 4 + i] = (__bf16)v[i];
      }
    }
#pragma unroll
    for (int j = 0; j < 4; ++j) {  // stage W tile transposed (32x128 -> Wt[n][k])
      int idx = tid + j * 256;
      int kr = idx >> 5, c4 = (idx & 31) * 4;
      v4f v = *(const v4f*)(W + (size_t)(kk + kr) * N + bN + c4);
#pragma unroll
      for (int i = 0; i < 4; ++i) Wt[c4 + i][kr] = (__bf16)v[i];
    }
    __syncthreads();

    v16bf a[2], b[4];
#pragma unroll
    for (int mt = 0; mt < 2; ++mt) a[mt] = load_a(&As[mW * 32 + mt * 16 + ln][0], h);
#pragma unroll
    for (int nt = 0; nt < 4; ++nt) b[nt] = load_b(&Wt[nW * 64 + nt * 16 + ln][0], h);
#pragma unroll
    for (int mt = 0; mt < 2; ++mt)
#pragma unroll
      for (int nt = 0; nt < 4; ++nt) acc[mt][nt] = WMMA_BF16(a[mt], b[nt], acc[mt][nt]);
    __syncthreads();
  }

  // C/D layout: elem r <-> row r+8h, col = lane&15
#pragma unroll
  for (int mt = 0; mt < 2; ++mt)
#pragma unroll
    for (int nt = 0; nt < 4; ++nt)
#pragma unroll
      for (int r = 0; r < 8; ++r) {
        int row = bM + mW * 32 + mt * 16 + 8 * h + r;
        int col = bN + nW * 64 + nt * 16 + ln;
        out[(size_t)row * N + col] = (__bf16)(acc[mt][nt][r] * scale);
      }
}

// ---------------------------------------------------------------------------
// Flash attention: block = 32 queries, KV tiles of 64, 8 waves.
// Wave w: mG=w>>2 (16-query group), dG=w&3 (16-key score group / 128-d O group)
// K tile is staged into LDS with global_load_async_to_lds_b128 (ASYNCcnt).
// ---------------------------------------------------------------------------
__global__ __launch_bounds__(256)
void flash_attn_kernel(const __bf16* __restrict__ Qb, const __bf16* __restrict__ Kb,
                       const __bf16* __restrict__ Vb, float* __restrict__ out) {
  constexpr int S = 2048, D = 512, TK = 64;
  constexpr int KS = 520;          // padded K-row stride (1040 B: 16B-aligned, 4-bank skew)
  __shared__ __bf16 Ks[TK][KS];    // K tile, row-major (async-staged)     66,560 B
  __shared__ __bf16 Vt[D][72];     // V tile transposed: Vt[d][k]          73,728 B
  __shared__ float  Sf[32][66];    // fp32 scores                           8,448 B
  __shared__ __bf16 Ps[32][72];    // exp(scores) bf16                      4,608 B
  __shared__ float  red[32][8];    // row-reduction scratch                 1,024 B
  __shared__ float  mrow[32], lrow[32], arow[32];

  const int tid  = threadIdx.x;
  const int lane = tid & 31, wv = tid >> 5;
  const int h = lane >> 4, ln = lane & 15;
  const int mG = wv >> 2, dG = wv & 3;
  const int bb = blockIdx.y;
  const int q0 = blockIdx.x * 32;

  const __bf16* Qbase = Qb + (size_t)bb * S * D;
  const __bf16* Kbase = Kb + (size_t)bb * S * D;
  const __bf16* Vbase = Vb + (size_t)bb * S * D;
  const unsigned long long kbase64 = (unsigned long long)(uintptr_t)Kbase;
  const unsigned ks_lds0 = (unsigned)(uintptr_t)&Ks[0][0];

  v8f oacc[8];
#pragma unroll
  for (int nt = 0; nt < 8; ++nt)
#pragma unroll
    for (int r = 0; r < 8; ++r) oacc[nt][r] = 0.0f;

  if (tid < 32) { mrow[tid] = -3.0e38f; lrow[tid] = 0.0f; arow[tid] = 0.0f; }
  __syncthreads();

  for (int kv0 = 0; kv0 < S; kv0 += TK) {
    // ---- kick async K-tile copy (64 x 512 bf16 = 64KB), 16B per lane-op.
    // Previous iteration's barriers guarantee all waves finished reading Ks.
#pragma unroll
    for (int j = 0; j < 16; ++j) {
      int idx = tid + j * 256;            // 0..4095 chunks of 16B
      int k   = idx >> 6;                 // key row 0..63
      int c   = idx & 63;                 // 16B chunk within row
      unsigned gvoff = (unsigned)((kv0 + k) * (D * 2) + c * 16);
      unsigned laddr = ks_lds0 + (unsigned)(k * (KS * 2) + c * 16);
      async_copy_b128(laddr, gvoff, kbase64);
    }

    // ---- stage V^T tile (64 x 512 bf16 -> Vt[d][k]) while the copy flies
    for (int j = 0; j < 32; ++j) {
      int idx = tid + j * 256;          // 0..8191
      int k   = idx >> 7;               // key 0..63
      int c4  = (idx & 127) * 4;        // d 0..508
      v4bf v = *(const v4bf*)(Vbase + (size_t)(kv0 + k) * D + c4);
#pragma unroll
      for (int i = 0; i < 4; ++i) Vt[c4 + i][k] = v[i];
    }
    if (kv0 + TK < S) {  // prefetch next V tile -> global_prefetch_b8
      const __bf16* pf = Vbase + (size_t)(kv0 + TK + (tid >> 2)) * D + (tid & 3) * 128;
      __builtin_prefetch(pf, 0, 1);
    }
    wait_asynccnt0();
    __syncthreads();

    // ---- scores: this wave's 16x16 tile of the 32x64 score block.
    // B = K^T, so per-lane B frag = contiguous run of LDS K row `ln`.
    {
      v8f sacc;
#pragma unroll
      for (int r = 0; r < 8; ++r) sacc[r] = 0.0f;
      const __bf16* qrow = Qbase + (size_t)(q0 + mG * 16 + ln) * D;
      const __bf16* krow = &Ks[dG * 16 + ln][0];
#pragma unroll
      for (int kk = 0; kk < D; kk += 32) {
        v16bf a = load_a(qrow + kk, h);
        v16bf b = load_b(krow + kk, h);
        sacc = WMMA_BF16(a, b, sacc);
      }
#pragma unroll
      for (int r = 0; r < 8; ++r) Sf[mG * 16 + 8 * h + r][dG * 16 + ln] = sacc[r];
    }
    __syncthreads();

    // ---- online softmax: thread t handles row t>>3, 8 cols at (t&7)*8
    {
      int r = tid >> 3, c = tid & 7;
      float mx = -3.0e38f;
#pragma unroll
      for (int j = 0; j < 8; ++j) mx = fmaxf(mx, Sf[r][c * 8 + j]);
      red[r][c] = mx;
      __syncthreads();
      if (c == 0) {
        float m2 = mrow[r];
#pragma unroll
        for (int j = 0; j < 8; ++j) m2 = fmaxf(m2, red[r][j]);
        arow[r] = __expf(mrow[r] - m2);
        mrow[r] = m2;
      }
      __syncthreads();
      float mcur = mrow[r], sum = 0.0f;
#pragma unroll
      for (int j = 0; j < 8; ++j) {
        float p = __expf(Sf[r][c * 8 + j] - mcur);
        Ps[r][c * 8 + j] = (__bf16)p;
        sum += p;
      }
      red[r][c] = sum;
      __syncthreads();
      if (c == 0) {
        float s2 = 0.0f;
#pragma unroll
        for (int j = 0; j < 8; ++j) s2 += red[r][j];
        lrow[r] = lrow[r] * arow[r] + s2;
      }
    }
    __syncthreads();

    // ---- rescale O, then O += P(16x64) @ V(64x128) for this wave's d-chunk
    {
      float al[8];
#pragma unroll
      for (int r = 0; r < 8; ++r) al[r] = arow[mG * 16 + 8 * h + r];
#pragma unroll
      for (int nt = 0; nt < 8; ++nt)
#pragma unroll
        for (int r = 0; r < 8; ++r) oacc[nt][r] *= al[r];

      v16bf pa[2];
#pragma unroll
      for (int kf = 0; kf < 2; ++kf) pa[kf] = load_a(&Ps[mG * 16 + ln][kf * 32], h);
#pragma unroll
      for (int nt = 0; nt < 8; ++nt) {
        const __bf16* vcol = &Vt[dG * 128 + nt * 16 + ln][0];
#pragma unroll
        for (int kf = 0; kf < 2; ++kf) {
          v16bf b = load_b(vcol + kf * 32, h);
          oacc[nt] = WMMA_BF16(pa[kf], b, oacc[nt]);
        }
      }
    }
    __syncthreads();
  }

  // ---- normalize by l and store fp32 output
  float li[8];
#pragma unroll
  for (int r = 0; r < 8; ++r) li[r] = 1.0f / lrow[mG * 16 + 8 * h + r];
#pragma unroll
  for (int nt = 0; nt < 8; ++nt)
#pragma unroll
    for (int r = 0; r < 8; ++r) {
      size_t row = (size_t)bb * S + q0 + mG * 16 + 8 * h + r;
      out[row * D + dG * 128 + nt * 16 + ln] = oacc[nt][r] * li[r];
    }
}

// ---------------------------------------------------------------------------
extern "C" void kernel_launch(void* const* d_in, const int* in_sizes, int n_in,
                              void* d_out, int out_size, void* d_ws, size_t ws_size,
                              hipStream_t stream) {
  (void)in_sizes; (void)n_in; (void)out_size; (void)ws_size;
  const float* x  = (const float*)d_in[0];
  const float* WQ = (const float*)d_in[1];
  const float* WK = (const float*)d_in[2];
  const float* WV = (const float*)d_in[3];
  float* out = (float*)d_out;

  const size_t per = (size_t)16384 * 512;   // B*S x D elements per tensor
  __bf16* Qb = (__bf16*)d_ws;
  __bf16* Kb = Qb + per;
  __bf16* Vb = Kb + per;                    // 48 MB bf16 workspace total

  const float scale = 0.04419417382415922f; // 1/sqrt(512), folded into Q

  dim3 gp(128, 4);  // 16384/128 x 512/128
  qkv_proj_kernel<<<gp, 256, 0, stream>>>(x, WQ, Qb, scale);
  qkv_proj_kernel<<<gp, 256, 0, stream>>>(x, WK, Kb, 1.0f);
  qkv_proj_kernel<<<gp, 256, 0, stream>>>(x, WV, Vb, 1.0f);

  dim3 ga(64, 8);   // S/32 query tiles x B batches
  flash_attn_kernel<<<ga, 256, 0, stream>>>(Qb, Kb, Vb, out);
}
